// Decoder_11476152615200
// MI455X (gfx1250) — matrix operand order repaired and checked by reading, not compiled, not run on hardware
//
#include <hip/hip_runtime.h>
#include <hip/hip_bf16.h>
#include <math.h>

// ---------------------------------------------------------------------------
// Types for CDNA5 WMMA (wave32): v16bf A/B operands, v8f f32 accumulator.
// ---------------------------------------------------------------------------
typedef __attribute__((ext_vector_type(16))) __bf16 v16bf;
typedef __attribute__((ext_vector_type(8)))  __bf16 v8bf;
typedef __attribute__((ext_vector_type(4)))  __bf16 v4bf;
typedef __attribute__((ext_vector_type(8)))  float  v8f;
typedef __attribute__((ext_vector_type(4)))  float  v4f;

// Problem constants
#define BB   64
#define TT   512
#define HID  1024
#define ENC  1024
#define EMB  512
#define VSZ  32000
#define KX   1536          // EMB+ENC (LSTM x width)
#define KZ   2560          // generic 2560-wide activation rows (xbuf/zbuf)

// Native bf16 convert -> v_cvt_pk_bf16_f32 on gfx1250 (RNE).
static __device__ __forceinline__ __bf16 f2bf(float f) { return (__bf16)f; }

// Branch-free fast transcendentals (v_exp_f32 based; TRANS ops co-issue with
// XDL WMMA per CDNA5 scheduling rules).
static __device__ __forceinline__ float fast_tanh(float x) {
  float e = __expf(2.0f * x);
  return 1.0f - __fdividef(2.0f, 1.0f + e);
}
static __device__ __forceinline__ float fast_sigm(float x) {
  return __fdividef(1.0f, 1.0f + __expf(-x));
}

// A operand: lane (m = lane&15, khalf = lane>>4) holds
// k = khalf*16 + {0..15} -> two contiguous 8-elt
// b128 loads (ds_load_b128 from LDS or global_load_b128 from memory; the
// helper is address-space generic and inlined). p = &row[khalf*16].
static __device__ __forceinline__ v16bf load_a16(const __bf16* p) {
  v8bf lo = *(const v8bf*)p;
  v8bf hi = *(const v8bf*)(p + 8);
  v16bf a;
#pragma unroll
  for (int i = 0; i < 8; ++i) { a[i] = lo[i]; a[i + 8] = hi[i]; }
  return a;
}

// B operand: lane owns column n = lane&15; element i maps to
// k = kbase + (lane>>4)*16 + i  ->  16 consecutive f32 of the weight row
// (weights stored [n][k] row-major). One 64B run per lane, packed-converted.
static __device__ __forceinline__ v16bf load_b_f32(const float* p) {
  v16bf b;
#pragma unroll
  for (int q = 0; q < 4; ++q) {
    v4f w = ((const v4f*)p)[q];
#pragma unroll
    for (int e = 0; e < 4; ++e) b[q * 4 + e] = f2bf(w[e]);
  }
  return b;
}

static __device__ __forceinline__ v8f wmma_bf16(v16bf a, v16bf b, v8f c) {
  return __builtin_amdgcn_wmma_f32_16x16x32_bf16(false, a, false, b, (short)0, c,
                                                 false, false);
}

// ---------------------------------------------------------------------------
// K0: build bf16 activation rows.
//   xbuf[b] = [ e(512) | ctx(1024 later) | h0(1024) ]   (LSTM GEMM A)
//   zbuf[b] = [ h_new(1024 later) | ctx(1024 later) | e(512) ]  (fc GEMM A)
// This kernel fills the embedding and h0 slices.
// ---------------------------------------------------------------------------
__global__ void k_embed(const int* __restrict__ tok,
                        const float* __restrict__ emb,
                        const float* __restrict__ h0,
                        __bf16* __restrict__ xbuf, __bf16* __restrict__ zbuf) {
  int b = blockIdx.x;
  int row = tok[b];
  for (int j = threadIdx.x; j < EMB; j += blockDim.x) {
    __bf16 v = f2bf(emb[(size_t)row * EMB + j]);
    xbuf[b * KZ + j] = v;
    zbuf[b * KZ + (HID + ENC) + j] = v;
  }
  for (int j = threadIdx.x; j < HID; j += blockDim.x) {
    xbuf[b * KZ + KX + j] = f2bf(h0[b * HID + j]);
  }
}

// ---------------------------------------------------------------------------
// K1: hb[b,h] = h0[b,:] . attn_W[h, 0:HID] + attn_b[h]   (tiny, VALU)
// ---------------------------------------------------------------------------
__global__ void k_hbias(const float* __restrict__ h0,
                        const float* __restrict__ attn_W,
                        const float* __restrict__ attn_b,
                        float* __restrict__ hb) {
  int b = blockIdx.x >> 2;
  int h = (blockIdx.x & 3) * 256 + threadIdx.x;
  const v4f* hr = (const v4f*)(h0 + (size_t)b * HID);
  const v4f* wr = (const v4f*)(attn_W + (size_t)h * (ENC + HID));
  float acc = attn_b[h];
#pragma unroll 4
  for (int k = 0; k < HID / 4; ++k) {
    v4f a = hr[k], w = wr[k];
    acc += a[0] * w[0] + a[1] * w[1] + a[2] * w[2] + a[3] * w[3];
  }
  hb[(size_t)b * HID + h] = acc;
}

// ---------------------------------------------------------------------------
// K2: fused attention scores.
// scores[bt] = sum_h v[h] * tanh( (enc_out @ We^T)[bt,h] + hb[b,h] )
// One WG = 64 consecutive bt rows (one batch b). enc_out rows staged once as
// bf16 in 128KB dynamic LDS (CDNA5: 320KB/WGP). 8 waves x 8 h-tiles, each
// tile = 4 row-tiles accumulated over K=1024 (32 wmma/tile).
// ---------------------------------------------------------------------------
__global__ void k_attn_scores(const float* __restrict__ enc_out,
                              const float* __restrict__ attn_W,
                              const float* __restrict__ hb,
                              const float* __restrict__ vW,
                              float* __restrict__ scores) {
  extern __shared__ __align__(16) char smem[];
  __bf16* Alds = (__bf16*)smem;                    // [64][1024] bf16 = 128KB
  float*  slds = (float*)(smem + 64 * ENC * 2);    // [64] f32 partial scores

  const int tid = threadIdx.x;
  const int bt0 = blockIdx.x * 64;
  const int b   = bt0 >> 9;                        // T = 512

  if (tid < 64) slds[tid] = 0.0f;

  // Stage A: 64 rows x 1024 cols, f32 -> bf16 (contiguous in enc_out)
  const v4f* src = (const v4f*)(enc_out + (size_t)bt0 * ENC);
  for (int i = tid; i < 64 * ENC / 4; i += 256) {
    v4f v = src[i];
    v4bf o;
#pragma unroll
    for (int e = 0; e < 4; ++e) o[e] = f2bf(v[e]);
    ((v4bf*)Alds)[i] = o;
  }
  __syncthreads();

  const int wave = tid >> 5, lane = tid & 31;
  const int row16 = lane & 15, khalf = lane >> 4;

  float spart[32];
#pragma unroll
  for (int i = 0; i < 32; ++i) spart[i] = 0.0f;

  for (int ht = wave; ht < 64; ht += 8) {
    int hcol = ht * 16 + row16;
    const float* wrow = attn_W + (size_t)hcol * (ENC + HID) + HID + khalf * 16;

    v8f acc[4];
#pragma unroll
    for (int rt = 0; rt < 4; ++rt)
#pragma unroll
      for (int e = 0; e < 8; ++e) acc[rt][e] = 0.0f;

    for (int ks = 0; ks < ENC; ks += 32) {
      v16bf bop = load_b_f32(wrow + ks);
#pragma unroll
      for (int rt = 0; rt < 4; ++rt) {
        const __bf16* ap = Alds + (rt * 16 + row16) * ENC + ks + khalf * 16;
        acc[rt] = wmma_bf16(load_a16(ap), bop, acc[rt]);
      }
    }
    float hbv = hb[(size_t)b * HID + hcol];
    float vv  = vW[hcol];
#pragma unroll
    for (int rt = 0; rt < 4; ++rt)
#pragma unroll
      for (int r = 0; r < 8; ++r)
        spart[rt * 8 + r] += vv * fast_tanh(acc[rt][r] + hbv);
  }
  // reduce across the 16 lanes holding each output row (ds_add_f32)
#pragma unroll
  for (int rt = 0; rt < 4; ++rt)
#pragma unroll
    for (int r = 0; r < 8; ++r)
      atomicAdd(&slds[rt * 16 + khalf * 8 + r], spart[rt * 8 + r]);
  __syncthreads();
  if (tid < 64) scores[bt0 + tid] = slds[tid];
}

// ---------------------------------------------------------------------------
// K3: softmax over T=512 per batch row, in place. 64 WGs x 256 threads.
// ---------------------------------------------------------------------------
__global__ void k_softmax(float* __restrict__ sc) {
  __shared__ float red[256];
  int b = blockIdx.x, tid = threadIdx.x;
  float s0 = sc[b * TT + tid];
  float s1 = sc[b * TT + 256 + tid];
  red[tid] = fmaxf(s0, s1);
  __syncthreads();
  for (int s = 128; s > 0; s >>= 1) {
    if (tid < s) red[tid] = fmaxf(red[tid], red[tid + s]);
    __syncthreads();
  }
  float M = red[0];
  __syncthreads();
  float e0 = __expf(s0 - M), e1 = __expf(s1 - M);
  red[tid] = e0 + e1;
  __syncthreads();
  for (int s = 128; s > 0; s >>= 1) {
    if (tid < s) red[tid] += red[tid + s];
    __syncthreads();
  }
  float inv = __fdividef(1.0f, red[0]);
  sc[b * TT + tid]       = e0 * inv;
  sc[b * TT + 256 + tid] = e1 * inv;
}

// ---------------------------------------------------------------------------
// K4: ctx[b,e] = sum_t w[b,t] * enc_out[b,t,e] -> bf16 slices of xbuf & zbuf.
// grid 256 (64 b x 4 col-chunks), coalesced over e.
// ---------------------------------------------------------------------------
__global__ void k_ctx(const float* __restrict__ wts,
                      const float* __restrict__ enc_out,
                      __bf16* __restrict__ xbuf, __bf16* __restrict__ zbuf) {
  int b = blockIdx.x >> 2;
  int e = (blockIdx.x & 3) * 256 + threadIdx.x;
  const float* wp = wts + b * TT;
  const float* ep = enc_out + ((size_t)b * TT) * ENC + e;
  float acc = 0.0f;
#pragma unroll 4
  for (int t = 0; t < TT; ++t) acc += wp[t] * ep[(size_t)t * ENC];
  __bf16 v = f2bf(acc);
  xbuf[b * KZ + EMB + e] = v;   // xbuf layout: [e | ctx | h0]
  zbuf[b * KZ + HID + e] = v;   // zbuf layout: [h_new | ctx | e]
}

// ---------------------------------------------------------------------------
// K5: gates = xbuf([e,ctx,h0]) @ [W_ih|W_hh]^T, fused LSTM cell.
// A is pre-converted bf16 in global (L2-resident) -> no LDS staging, no
// barriers in the K loop. WG w owns hid cols [w*16,w*16+16) across all 4
// gates; i/f/g/o meet in LDS once at the end.
// ---------------------------------------------------------------------------
__global__ void k_gates_cell(const __bf16* __restrict__ xbuf,
                             const float* __restrict__ c0,
                             const float* __restrict__ W_ih,
                             const float* __restrict__ W_hh,
                             const float* __restrict__ b_ih,
                             const float* __restrict__ b_hh,
                             float* __restrict__ out_h,
                             float* __restrict__ out_c,
                             __bf16* __restrict__ zbuf) {
  __shared__ float glds[4 * 64 * 16];                // 16 KB gate tiles

  const int tid = threadIdx.x;
  const int hc0 = blockIdx.x * 16;
  const int wave = tid >> 5, lane = tid & 31;
  const int row16 = lane & 15, khalf = lane >> 4;
  const int m = wave & 3, gpair = wave >> 2;

  v8f acc[2];
#pragma unroll
  for (int s = 0; s < 2; ++s)
#pragma unroll
    for (int e = 0; e < 8; ++e) acc[s][e] = 0.0f;

  const __bf16* arow = xbuf + (size_t)(m * 16 + row16) * KZ + khalf * 16;
  for (int ks = 0; ks < KZ; ks += 32) {
    int kk0 = ks + khalf * 16;
    v16bf aop = load_a16(arow + ks);
#pragma unroll
    for (int s = 0; s < 2; ++s) {
      int n = (gpair * 2 + s) * HID + hc0 + row16;
      const float* bp = (kk0 < KX) ? (W_ih + (size_t)n * KX + kk0)
                                   : (W_hh + (size_t)n * HID + (kk0 - KX));
      acc[s] = wmma_bf16(aop, load_b_f32(bp), acc[s]);
    }
  }
#pragma unroll
  for (int s = 0; s < 2; ++s)
#pragma unroll
    for (int r = 0; r < 8; ++r)
      glds[((gpair * 2 + s) * 64 + m * 16 + khalf * 8 + r) * 16 + row16] =
          acc[s][r];
  __syncthreads();

  for (int idx = tid; idx < 1024; idx += 256) {
    int row = idx >> 4, col = idx & 15, hcol = hc0 + col;
    float gi = glds[(0 * 64 + row) * 16 + col] + b_ih[hcol] + b_hh[hcol];
    float gf = glds[(1 * 64 + row) * 16 + col] + b_ih[HID + hcol] + b_hh[HID + hcol];
    float gg = glds[(2 * 64 + row) * 16 + col] + b_ih[2 * HID + hcol] + b_hh[2 * HID + hcol];
    float go = glds[(3 * 64 + row) * 16 + col] + b_ih[3 * HID + hcol] + b_hh[3 * HID + hcol];
    float iv = fast_sigm(gi), fv = fast_sigm(gf), ov = fast_sigm(go);
    float gv = fast_tanh(gg);
    float cn = fv * c0[row * HID + hcol] + iv * gv;
    float hn = ov * fast_tanh(cn);
    out_h[row * HID + hcol] = hn;
    out_c[row * HID + hcol] = cn;
    zbuf[row * KZ + hcol] = f2bf(hn);
  }
}

// ---------------------------------------------------------------------------
// K6: pred = zbuf[64,2560] @ fc_W^T + fc_b.  500 WGs x 64 output columns.
// fc_W (327 MB) read exactly once -> bandwidth-limited; prefetch the weight
// stream (global_prefetch_b8) and keep ALU free via WMMA.
// ---------------------------------------------------------------------------
__global__ void k_fc(const __bf16* __restrict__ zbuf,
                     const float* __restrict__ fc_W,
                     const float* __restrict__ fc_b,
                     float* __restrict__ pred) {
  const int tid = threadIdx.x;
  const int n0 = blockIdx.x * 64;
  const int wave = tid >> 5, lane = tid & 31;
  const int row16 = lane & 15, khalf = lane >> 4;
  const int m = wave & 3, npair = wave >> 2;

  v8f acc[2];
#pragma unroll
  for (int s = 0; s < 2; ++s)
#pragma unroll
    for (int e = 0; e < 8; ++e) acc[s][e] = 0.0f;

  const __bf16* arow = zbuf + (size_t)(m * 16 + row16) * KZ + khalf * 16;
  const float* brow[2];
#pragma unroll
  for (int s = 0; s < 2; ++s) {
    int n = n0 + (npair * 2 + s) * 16 + row16;
    brow[s] = fc_W + (size_t)n * KZ + khalf * 16;
  }

  for (int ks = 0; ks < KZ; ks += 32) {
    v16bf aop = load_a16(arow + ks);
#pragma unroll
    for (int s = 0; s < 2; ++s) {
      __builtin_prefetch(brow[s] + ks + 64, 0, 0);   // next 256B of the stream
      acc[s] = wmma_bf16(aop, load_b_f32(brow[s] + ks), acc[s]);
    }
  }
#pragma unroll
  for (int s = 0; s < 2; ++s) {
    int ncol = n0 + (npair * 2 + s) * 16 + row16;
    float bias = fc_b[ncol];
#pragma unroll
    for (int r = 0; r < 8; ++r) {
      int row = m * 16 + khalf * 8 + r;
      pred[(size_t)row * VSZ + ncol] = acc[s][r] + bias;
    }
  }
}

// ---------------------------------------------------------------------------
// Host-side orchestration
// ---------------------------------------------------------------------------
extern "C" void kernel_launch(void* const* d_in, const int* in_sizes, int n_in,
                              void* d_out, int out_size, void* d_ws, size_t ws_size,
                              hipStream_t stream) {
  const int*   tok      = (const int*)  d_in[0];
  const float* h        = (const float*)d_in[1];   // [1,B,HID]
  const float* c        = (const float*)d_in[2];   // [1,B,HID]
  const float* enc_out  = (const float*)d_in[3];   // [B,T,ENC]
  const float* emb      = (const float*)d_in[4];   // [VSZ,EMB]
  const float* attn_W   = (const float*)d_in[5];   // [HID, ENC+HID]
  const float* attn_b   = (const float*)d_in[6];   // [HID]
  const float* v_W      = (const float*)d_in[7];   // [1,HID]
  const float* W_ih     = (const float*)d_in[8];   // [4H, KX]
  const float* W_hh     = (const float*)d_in[9];   // [4H, HID]
  const float* b_ih     = (const float*)d_in[10];  // [4H]
  const float* b_hh     = (const float*)d_in[11];  // [4H]
  const float* fc_W     = (const float*)d_in[12];  // [VSZ, KZ]
  const float* fc_b     = (const float*)d_in[13];  // [VSZ]

  char* ws = (char*)d_ws;
  float*  scores = (float*)ws;                                   // 32768 f32
  float*  hb     = (float*)(ws + 131072);                        // 65536 f32
  __bf16* xbuf   = (__bf16*)(ws + 393216);                       // 64x2560 bf16
  __bf16* zbuf   = (__bf16*)(ws + 393216 + (size_t)BB * KZ * 2); // 64x2560 bf16
                                                                 // ~1.0 MB total

  float* out    = (float*)d_out;
  float* pred   = out;                         // [B, VSZ]
  float* out_h  = out + (size_t)BB * VSZ;      // [1,B,HID]
  float* out_c  = out_h + (size_t)BB * HID;    // [1,B,HID]

  k_embed<<<BB, 256, 0, stream>>>(tok, emb, h, xbuf, zbuf);
  k_hbias<<<256, 256, 0, stream>>>(h, attn_W, attn_b, hb);

  size_t attn_smem = (size_t)64 * ENC * sizeof(__bf16) + 64 * sizeof(float);
  k_attn_scores<<<(BB * TT) / 64, 256, attn_smem, stream>>>(
      enc_out, attn_W, hb, v_W, scores);

  k_softmax<<<BB, 256, 0, stream>>>(scores);
  k_ctx<<<BB * 4, 256, 0, stream>>>(scores, enc_out, xbuf, zbuf);
  k_gates_cell<<<HID / 16, 256, 0, stream>>>(xbuf, c, W_ih, W_hh, b_ih, b_hh,
                                             out_h, out_c, zbuf);
  k_fc<<<VSZ / 64, 256, 0, stream>>>(zbuf, fc_W, fc_b, pred);
}
